// InnerShiftTriple_17291538333934
// MI455X (gfx1250) — matrix operand order repaired and checked by reading, not compiled, not run on hardware
//
#include <hip/hip_runtime.h>
#include <hip/hip_fp16.h>

// ---- problem constants (from reference) ----
#define BZV   4
#define CFULL 512
#define C2    256
#define HWV   4096      // 64*64
#define KH    8         // 256 / 32 K-steps per 16x16 tile
#define NWAVE 4         // key-split waves per workgroup
#define KEYS_PER_WAVE (HWV / NWAVE)   // 1024

typedef _Float16 v8h  __attribute__((ext_vector_type(8)));
typedef _Float16 v16h __attribute__((ext_vector_type(16)));
typedef float    v8f  __attribute__((ext_vector_type(8)));

// Load 16 contiguous halfs (32B) as a v16h fragment (two b128 loads).
static __device__ __forceinline__ v16h load_frag(const _Float16* __restrict__ p) {
    v8h lo = *(const v8h*)(p);
    v8h hp = *(const v8h*)(p + 8);
    v16h f;
#pragma unroll
    for (int t = 0; t < 8; ++t) { f[t] = lo[t]; f[8 + t] = hp[t]; }
    return f;
}

static __device__ __forceinline__ v8f wmma_f16(v16h a, v16h b, v8f c) {
    return __builtin_amdgcn_wmma_f32_16x16x32_f16(
        false, a, false, b, (short)0, c, false, false);
}

// -----------------------------------------------------------------------------
// Kernel 1: per-pixel L2-normalize the "latter" half into f16, layout
// lat16[b][pixel][k] (K contiguous, 256 halfs = 512B per row) so WMMA
// fragment loads are contiguous 16B chunks per lane.
// -----------------------------------------------------------------------------
__global__ void k_normalize(const float* __restrict__ in,
                            _Float16* __restrict__ lat16) {
    const int b = blockIdx.y;
    const int i = blockIdx.x * blockDim.x + threadIdx.x;   // pixel index
    const float* __restrict__ latter = in + ((size_t)b * CFULL + C2) * HWV;

    float ss = 0.0f;
#pragma unroll 8
    for (int c = 0; c < C2; ++c) {
        float v = latter[(size_t)c * HWV + i];
        ss += v * v;
    }
    const float inv = 1.0f / (sqrtf(ss) + 1e-8f);

    _Float16* __restrict__ dst = lat16 + ((size_t)b * HWV + i) * C2;
#pragma unroll 2
    for (int c0 = 0; c0 < C2; c0 += 8) {
        v8h pack;
#pragma unroll
        for (int t = 0; t < 8; ++t)
            pack[t] = (_Float16)(latter[(size_t)(c0 + t) * HWV + i] * inv);
        *(v8h*)(dst + c0) = pack;
    }
}

// -----------------------------------------------------------------------------
// Kernel 2: fused Gram-matrix (sim = L^T L) + masked argmax over keys.
//
// Block = 128 threads (4 waves). The block owns a 32-query-row panel; each
// wave scans a disjoint quarter of the 4096 keys. Per wave: A tiles (32x256)
// live in 128 VGPRs for the whole key loop; per 16-key tile one shared set of
// B fragment loads feeds 16 WMMAs arranged as FOUR independent accumulator
// chains (2 query tiles x even/odd K split) for XDL ILP. launch_bounds(.,1)
// grants the full VGPR budget so nothing spills.
//
// C/D layout (ISA 7.12.2): VGPR v, lane L -> row = v + 8*(L>>4), col = L&15.
// Per-lane running (max,argmax); shfl_xor reduction within each 16-lane half;
// final cross-wave combine through LDS with first-index tie-breaking (wave
// key ranges are disjoint & increasing, so argmax-first semantics hold).
// -----------------------------------------------------------------------------
__global__ void
__launch_bounds__(32 * NWAVE, 1)
k_sim_argmax(const _Float16* __restrict__ lat16,
             const int* __restrict__ mask,
             int* __restrict__ srcidx) {
    const int b    = blockIdx.y;
    const int i0   = blockIdx.x * 32;          // query panel start (32 rows)
    const int wave = threadIdx.x >> 5;         // 0..3
    const int L    = threadIdx.x & 31;         // lane 0..31
    const int lm   = L & 15;
    const int hi   = L >> 4;

    __shared__ float ldsVal[NWAVE][32];
    __shared__ int   ldsIdx[NWAVE][32];

    const _Float16* __restrict__ base = lat16 + (size_t)b * HWV * C2;

    // ---- load A fragments for both query tiles:
    // row m = i0 + lm (+16 for tile 1); halfs[0..7] = K(ks*32 + hi*16 + t),
    // halfs[8..15] = K(ks*32 + hi*16 + 8 + t) (contiguous, matches B)
    v16h a0[KH], a1[KH];
    {
        const _Float16* __restrict__ arow0 = base + (size_t)(i0 + lm) * C2;
        const _Float16* __restrict__ arow1 = arow0 + (size_t)16 * C2;
#pragma unroll
        for (int ks = 0; ks < KH; ++ks) {
            const int o0 = ks * 32 + hi * 16;
            const int o1 = ks * 32 + hi * 16 + 8;
            v8h l0 = *(const v8h*)(arow0 + o0);
            v8h h0 = *(const v8h*)(arow0 + o1);
            v8h l1 = *(const v8h*)(arow1 + o0);
            v8h h1 = *(const v8h*)(arow1 + o1);
            v16h f0, f1;
#pragma unroll
            for (int t = 0; t < 8; ++t) {
                f0[t] = l0[t]; f0[8 + t] = h0[t];
                f1[t] = l1[t]; f1[8 + t] = h1[t];
            }
            a0[ks] = f0;
            a1[ks] = f1;
        }
    }

    float best0[8], best1[8];
    int   bid0[8],  bid1[8];
#pragma unroll
    for (int v = 0; v < 8; ++v) {
        best0[v] = -3.0e38f; bid0[v] = 0;
        best1[v] = -3.0e38f; bid1[v] = 0;
    }

    const int* __restrict__ mrow = mask + (size_t)b * HWV;

    const int jBeg = wave * KEYS_PER_WAVE;
    const int jEnd = jBeg + KEYS_PER_WAVE;
    for (int j0 = jBeg; j0 < jEnd; j0 += 16) {
        const int  key  = j0 + lm;           // this lane's key column
        const bool hole = (mrow[key] >= 1);  // invalid shift source

        // ---- B fragments: col n = key, halfs[0..15] = K(ks*32 + hi*16 + t)
        const _Float16* __restrict__ brow = base + (size_t)key * C2 + hi * 16;

        v8f acc0a = {}, acc0b = {}, acc1a = {}, acc1b = {};
#pragma unroll
        for (int ks = 0; ks < KH; ks += 2) {
            v16h bf0 = load_frag(brow + ks * 32);
            v16h bf1 = load_frag(brow + (ks + 1) * 32);
            acc0a = wmma_f16(a0[ks],     bf0, acc0a);
            acc1a = wmma_f16(a1[ks],     bf0, acc1a);
            acc0b = wmma_f16(a0[ks + 1], bf1, acc0b);
            acc1b = wmma_f16(a1[ks + 1], bf1, acc1b);
        }

        // masked keys contribute -1e9 (reference semantics); strict '>' with
        // monotonically increasing j keeps the first max index.
#pragma unroll
        for (int v = 0; v < 8; ++v) {
            const float s0 = acc0a[v] + acc0b[v];
            const float s1 = acc1a[v] + acc1b[v];
            const float v0 = hole ? -1.0e9f : s0;
            const float v1 = hole ? -1.0e9f : s1;
            if (v0 > best0[v]) { best0[v] = v0; bid0[v] = key; }
            if (v1 > best1[v]) { best1[v] = v1; bid1[v] = key; }
        }
    }

    // ---- per-row reduction across the 16 lanes of each half-wave ----
#pragma unroll
    for (int v = 0; v < 8; ++v) {
        float val0 = best0[v]; int idx0 = bid0[v];
        float val1 = best1[v]; int idx1 = bid1[v];
#pragma unroll
        for (int s = 8; s >= 1; s >>= 1) {
            float ov0 = __shfl_xor(val0, s, 32);
            int   oi0 = __shfl_xor(idx0, s, 32);
            float ov1 = __shfl_xor(val1, s, 32);
            int   oi1 = __shfl_xor(idx1, s, 32);
            if (ov0 > val0 || (ov0 == val0 && oi0 < idx0)) { val0 = ov0; idx0 = oi0; }
            if (ov1 > val1 || (ov1 == val1 && oi1 < idx1)) { val1 = ov1; idx1 = oi1; }
        }
        if (lm == 0) {
            const int row = v + 8 * hi;          // 0..15
            ldsVal[wave][row]      = val0; ldsIdx[wave][row]      = idx0;
            ldsVal[wave][16 + row] = val1; ldsIdx[wave][16 + row] = idx1;
        }
    }

    __syncthreads();

    // ---- cross-wave combine: one lane per panel row ----
    if (threadIdx.x < 32) {
        const int row = threadIdx.x;             // 0..31 -> global row i0+row
        float bv = ldsVal[0][row];
        int   bi = ldsIdx[0][row];
#pragma unroll
        for (int w = 1; w < NWAVE; ++w) {
            const float ov = ldsVal[w][row];
            const int   oi = ldsIdx[w][row];
            if (ov > bv || (ov == bv && oi < bi)) { bv = ov; bi = oi; }
        }
        srcidx[(size_t)b * HWV + i0 + row] = bi;
    }
}

// -----------------------------------------------------------------------------
// Kernel 3: output assembly. out[b] = concat(former, latter, shifted) along C.
// Pure bandwidth (~80 MB total traffic); gather source is L2-resident (16 MB).
// -----------------------------------------------------------------------------
__global__ void k_assemble(const float* __restrict__ in,
                           const int* __restrict__ mask,
                           const int* __restrict__ srcidx,
                           float* __restrict__ out) {
    const size_t total = (size_t)BZV * 768 * HWV;
    size_t e = (size_t)blockIdx.x * blockDim.x + threadIdx.x;
    if (e >= total) return;

    const int i = (int)(e % HWV);
    const int c = (int)((e / HWV) % 768);
    const int b = (int)(e / ((size_t)768 * HWV));

    if (c < CFULL) {
        // former + latter pass-through
        out[e] = in[((size_t)b * CFULL + c) * HWV + i];
    } else {
        const int  cc   = c - CFULL;                      // 0..255 -> former chan
        const bool hole = (mask[(size_t)b * HWV + i] >= 1);
        float v = 0.0f;
        if (hole) {
            const int idx = srcidx[(size_t)b * HWV + i];
            v = in[((size_t)b * CFULL + cc) * HWV + idx]; // gather from former
        }
        out[e] = v;
    }
}

// -----------------------------------------------------------------------------
extern "C" void kernel_launch(void* const* d_in, const int* in_sizes, int n_in,
                              void* d_out, int out_size, void* d_ws, size_t ws_size,
                              hipStream_t stream) {
    const float* input = (const float*)d_in[0];   // [4,512,64,64] f32
    const int*   mask  = (const int*)d_in[1];     // [4,1,64,64] i32
    float*       out   = (float*)d_out;           // [4,768,64,64] f32

    _Float16* lat16  = (_Float16*)d_ws;                                   // 8 MB
    int*      srcidx = (int*)((char*)d_ws + (size_t)BZV * HWV * C2 * 2);  // 64 KB

    // 1) normalize latter -> f16 [b][pixel][k]
    dim3 g1(HWV / 256, BZV);
    k_normalize<<<g1, 256, 0, stream>>>(input, lat16);

    // 2) WMMA Gram matrix + masked argmax
    //    (128-thread blocks: 32 query rows x 4 key-split waves)
    dim3 g2(HWV / 32, BZV);
    k_sim_argmax<<<g2, 32 * NWAVE, 0, stream>>>(lat16, mask, srcidx);

    // 3) assemble output
    const size_t total = (size_t)BZV * 768 * HWV;
    k_assemble<<<(unsigned)((total + 255) / 256), 256, 0, stream>>>(
        input, mask, srcidx, out);
}
